// LIIF_Sampler_48009144435068
// MI455X (gfx1250) — matrix operand order repaired and checked by reading, not compiled, not run on hardware
//
#include <hip/hip_runtime.h>

// ---------------- problem constants ----------------
#define NB   4
#define C_IN 64
#define HF   48
#define WF   48
#define HQ   384
#define WQ   384
#define NPTS   (NB*HQ*WQ)      // 589824 query points
#define NTILES (NPTS/16)       // 36864 16-point tiles
#define NPAIRS (NTILES/2)      // 18432 tile pairs (2 tiles per wave)
#define ROWS   272             // ushort stride of one LDS activation row (544B)

// packed-weight offsets in workspace (ushort units)
#define OFF0 0                          // 96x256  -> 3*16 frags * 512
#define OFF1 (OFF0 + 3*16*512)          // 256x256 -> 8*16 frags
#define OFF2 (OFF1 + 8*16*512)
#define OFF3 (OFF2 + 8*16*512)
#define OFF4 (OFF3 + 8*16*512)          // 256x16 (3 valid cols) -> 8 frags
#define WP_TOTAL (OFF4 + 8*1*512)       // 225280 ushorts = 440KB

typedef __attribute__((ext_vector_type(16))) __bf16 v16bf;
typedef __attribute__((ext_vector_type(8)))  float  v8f;

union AFrag { v16bf v; uint4 q[2]; };

__device__ __forceinline__ unsigned short f2bf(float f) {
  unsigned u = __builtin_bit_cast(unsigned, f);
  u += 0x7fffu + ((u >> 16) & 1u);      // round-to-nearest-even
  return (unsigned short)(u >> 16);
}

// ---------------------------------------------------------------------------
// Pre-swizzle one weight matrix (fp32 [K x ld], valid cols < Nvalid) into
// WMMA B-fragment order, bf16. Fragment f = ks*Ntiles + nt is 32 lanes x 16
// bf16; lane (h = lane>>4, c = lane&15) element e holds B[ks*32 + 16h + e][nt*16+c].
// ---------------------------------------------------------------------------
__global__ void pack_w(const float* __restrict__ W, int K, int Kpad, int ld,
                       int Nvalid, int Ntiles, unsigned short* __restrict__ dst) {
  int t = blockIdx.x * blockDim.x + threadIdx.x;
  int total = (Kpad / 32) * Ntiles * 32;
  if (t >= total) return;
  int lane = t & 31, f = t >> 5;
  int ks = f / Ntiles, nt = f % Ntiles;
  int h = lane >> 4, c = lane & 15, col = nt * 16 + c;
  unsigned short* o = dst + (size_t)f * 512 + lane * 16;
#pragma unroll
  for (int e = 0; e < 16; ++e) {
    int k = ks * 32 + 16 * h + e;
    float v = (k < K && col < Nvalid) ? W[(size_t)k * ld + col] : 0.f;
    o[e] = f2bf(v);
  }
}

// ---------------------------------------------------------------------------
// Featurize one 16-point tile into an LDS activation slab (bf16, K padded 96).
// ---------------------------------------------------------------------------
__device__ __forceinline__ void featurize(
    int tile, unsigned short (*__restrict__ act)[ROWS], int* __restrict__ meta,
    const float* __restrict__ xfeat, const float* __restrict__ coord,
    const float* __restrict__ cell,  const float* __restrict__ lr,
    int lane, int h, int rc) {
  if (h == 0) {                       // lanes 0..15: one point each
    const int p = tile * 16 + rc;
    const int b = p / (HQ * WQ);
    const float gy = coord[(size_t)p * 2 + 0];
    const float gx = coord[(size_t)p * 2 + 1];
    float fx = ((gx + 1.f) * WF - 1.f) * 0.5f;
    float fy = ((gy + 1.f) * HF - 1.f) * 0.5f;
    int ixn = (int)rintf(fx), iyn = (int)rintf(fy);
    int valid = (ixn >= 0 && ixn < WF && iyn >= 0 && iyn < HF) ? 1 : 0;
    int ixc = min(max(ixn, 0), WF - 1), iyc = min(max(iyn, 0), HF - 1);
    meta[rc] = (valid << 16) | (iyc << 8) | ixc;
    float qcy = valid ? ((2.f * iyc + 1.f) / HF - 1.f) : 0.f;
    float qcx = valid ? ((2.f * ixc + 1.f) / WF - 1.f) : 0.f;
    unsigned short* row = act[rc];
    row[64] = f2bf((gy - qcy) * HF);
    row[65] = f2bf((gx - qcx) * WF);
    row[66] = f2bf(cell[b * 2 + 0] * HF);
    row[67] = f2bf(cell[b * 2 + 1] * WF);
    // rgb std over 4 diagonally shifted nearest LR samples (ddof=1)
    float s1[3] = {0, 0, 0}, s2[3] = {0, 0, 0};
    const float rx = 1.f / HF, ry = 1.f / WF, eps = 1e-6f;
    const float lo = -1.f + 1e-6f, hi = 1.f - 1e-6f;
#pragma unroll
    for (int vi = 0; vi < 4; ++vi) {
      float vx = (vi < 2) ? -1.f : 1.f;
      float vy = (vi & 1) ? 1.f : -1.f;
      float cy = fminf(fmaxf(gy + vx * rx + eps, lo), hi);
      float cx = fminf(fmaxf(gx + vy * ry + eps, lo), hi);
      int jx = (int)rintf(((cx + 1.f) * WF - 1.f) * 0.5f);
      int jy = (int)rintf(((cy + 1.f) * HF - 1.f) * 0.5f);
      int v2 = (jx >= 0 && jx < WF && jy >= 0 && jy < HF);
      int jxc = min(max(jx, 0), WF - 1), jyc = min(max(jy, 0), HF - 1);
#pragma unroll
      for (int ch = 0; ch < 3; ++ch) {
        float s = v2 ? lr[(((size_t)b * 3 + ch) * HF + jyc) * WF + jxc] : 0.f;
        s1[ch] += s; s2[ch] += s * s;
      }
    }
#pragma unroll
    for (int ch = 0; ch < 3; ++ch) {
      float mean = s1[ch] * 0.25f;
      float var  = fmaxf((s2[ch] - 4.f * mean * mean) * (1.f / 3.f), 0.f);
      row[68 + ch] = f2bf(sqrtf(var));
    }
#pragma unroll
    for (int k = 71; k < 96; ++k) row[k] = 0;   // zero-pad K to 96
  }
  // 16 rows x 64 feature channels, all 32 lanes
  for (int i = lane; i < 16 * 64; i += 32) {
    int r = i >> 6, ch = i & 63;
    int m = meta[r];
    int p = tile * 16 + r;
    int b = p / (HQ * WQ);
    float v = 0.f;
    if (m & (1 << 16)) {
      int iy = (m >> 8) & 255, ix = m & 255;
      v = xfeat[(((size_t)b * C_IN + ch) * HF + iy) * WF + ix];
    }
    act[r][ch] = f2bf(v);
  }
}

// ---------------------------------------------------------------------------
// One MLP layer (ReLU) over TWO 16-point tiles: every B-fragment load feeds
// two WMMAs. KS = K/32 steps (3 for layer 0, 8 for hidden layers).
// ---------------------------------------------------------------------------
template <int KS>
__device__ __forceinline__ void mlp_layer(
    unsigned short (*__restrict__ actA)[ROWS],
    unsigned short (*__restrict__ actB)[ROWS],
    const unsigned short* __restrict__ wl, const float* __restrict__ bl,
    int lane, int h, int rc) {
  AFrag a[2][KS];
#pragma unroll
  for (int ks = 0; ks < KS; ++ks) {
    a[0][ks].q[0] = *(const uint4*)&actA[rc][ks * 32 + 16 * h];
    a[0][ks].q[1] = *(const uint4*)&actA[rc][ks * 32 + 8 + 16 * h];
    a[1][ks].q[0] = *(const uint4*)&actB[rc][ks * 32 + 16 * h];
    a[1][ks].q[1] = *(const uint4*)&actB[rc][ks * 32 + 8 + 16 * h];
  }
  for (int nt = 0; nt < 16; ++nt) {
    v8f acc0 = {}, acc1 = {};
#pragma unroll
    for (int ks = 0; ks < KS; ++ks) {
      AFrag bfr;
      const uint4* pB = (const uint4*)(wl + (size_t)(ks * 16 + nt) * 512 + lane * 16);
      bfr.q[0] = pB[0]; bfr.q[1] = pB[1];
      acc0 = __builtin_amdgcn_wmma_f32_16x16x32_bf16(
          false, a[0][ks].v, false, bfr.v, (short)0, acc0, false, false);
      acc1 = __builtin_amdgcn_wmma_f32_16x16x32_bf16(
          false, a[1][ks].v, false, bfr.v, (short)0, acc1, false, false);
    }
    float bb = bl[nt * 16 + rc];
#pragma unroll
    for (int j = 0; j < 8; ++j) {
      actA[j + 8 * h][nt * 16 + rc] = f2bf(fmaxf(acc0[j] + bb, 0.f));
      actB[j + 8 * h][nt * 16 + rc] = f2bf(fmaxf(acc1[j] + bb, 0.f));
    }
  }
}

// ---------------------------------------------------------------------------
// Fused featurize + 5-layer MLP; one wave = two 16-point tiles.
// ---------------------------------------------------------------------------
__global__ void __launch_bounds__(64) liif_fused(
    const float* __restrict__ xfeat, const float* __restrict__ coord,
    const float* __restrict__ cell,  const float* __restrict__ lr,
    const float* __restrict__ b0, const float* __restrict__ b1,
    const float* __restrict__ b2, const float* __restrict__ b3,
    const float* __restrict__ b4,
    const unsigned short* __restrict__ wp, float* __restrict__ out) {
  __shared__ __align__(16) unsigned short s_act[2][2][16][ROWS];   // [wave][slab]
  __shared__ int s_meta[2][2][16];

  const int wave = threadIdx.x >> 5;
  const int lane = threadIdx.x & 31;
  const int h  = lane >> 4;       // K-half (A/B) / M-half (C/D)
  const int rc = lane & 15;       // A row / B,C,D column
  unsigned short(*act0)[ROWS] = s_act[wave][0];
  unsigned short(*act1)[ROWS] = s_act[wave][1];
  int* meta0 = s_meta[wave][0];
  int* meta1 = s_meta[wave][1];

  const int gwave  = blockIdx.x * 2 + wave;
  const int nwaves = gridDim.x * 2;

  for (int pair = gwave; pair < NPAIRS; pair += nwaves) {
    const int t0 = pair * 2, t1 = pair * 2 + 1;
    featurize(t0, act0, meta0, xfeat, coord, cell, lr, lane, h, rc);
    featurize(t1, act1, meta1, xfeat, coord, cell, lr, lane, h, rc);

    mlp_layer<3>(act0, act1, wp + OFF0, b0, lane, h, rc);   // 96  -> 256
    mlp_layer<8>(act0, act1, wp + OFF1, b1, lane, h, rc);   // 256 -> 256
    mlp_layer<8>(act0, act1, wp + OFF2, b2, lane, h, rc);
    mlp_layer<8>(act0, act1, wp + OFF3, b3, lane, h, rc);

    // ---- head: 16x256 @ 256x16 (3 valid cols), fp32 scatter to [B,3,HQ,WQ]
    AFrag a[2][8];
#pragma unroll
    for (int ks = 0; ks < 8; ++ks) {
      a[0][ks].q[0] = *(const uint4*)&act0[rc][ks * 32 + 16 * h];
      a[0][ks].q[1] = *(const uint4*)&act0[rc][ks * 32 + 8 + 16 * h];
      a[1][ks].q[0] = *(const uint4*)&act1[rc][ks * 32 + 16 * h];
      a[1][ks].q[1] = *(const uint4*)&act1[rc][ks * 32 + 8 + 16 * h];
    }
    v8f acc0 = {}, acc1 = {};
#pragma unroll
    for (int ks = 0; ks < 8; ++ks) {
      AFrag bfr;
      const uint4* pB = (const uint4*)(wp + OFF4 + (size_t)ks * 512 + lane * 16);
      bfr.q[0] = pB[0]; bfr.q[1] = pB[1];
      acc0 = __builtin_amdgcn_wmma_f32_16x16x32_bf16(
          false, a[0][ks].v, false, bfr.v, (short)0, acc0, false, false);
      acc1 = __builtin_amdgcn_wmma_f32_16x16x32_bf16(
          false, a[1][ks].v, false, bfr.v, (short)0, acc1, false, false);
    }
    if (rc < 3) {
      float bb = b4[rc];
#pragma unroll
      for (int j = 0; j < 8; ++j) {
        int p = t0 * 16 + j + 8 * h;
        int b = p / (HQ * WQ);
        out[((size_t)b * 3 + rc) * (HQ * WQ) + (p - b * (HQ * WQ))] = acc0[j] + bb;
        p = t1 * 16 + j + 8 * h;
        b = p / (HQ * WQ);
        out[((size_t)b * 3 + rc) * (HQ * WQ) + (p - b * (HQ * WQ))] = acc1[j] + bb;
      }
    }
  }
}

extern "C" void kernel_launch(void* const* d_in, const int* in_sizes, int n_in,
                              void* d_out, int out_size, void* d_ws, size_t ws_size,
                              hipStream_t stream) {
  const float* x     = (const float*)d_in[0];
  const float* coord = (const float*)d_in[1];
  const float* cell  = (const float*)d_in[2];
  const float* lr    = (const float*)d_in[3];
  const float* W0 = (const float*)d_in[4];  const float* b0 = (const float*)d_in[5];
  const float* W1 = (const float*)d_in[6];  const float* b1 = (const float*)d_in[7];
  const float* W2 = (const float*)d_in[8];  const float* b2 = (const float*)d_in[9];
  const float* W3 = (const float*)d_in[10]; const float* b3 = (const float*)d_in[11];
  const float* W4 = (const float*)d_in[12]; const float* b4 = (const float*)d_in[13];
  unsigned short* wp = (unsigned short*)d_ws;   // needs WP_TOTAL*2 = 440KB

  // pre-swizzle weights into WMMA B-fragment order (bf16), L2-resident
  pack_w<<<(3 * 16 * 32 + 255) / 256, 256, 0, stream>>>(W0,  71,  96, 256, 256, 16, wp + OFF0);
  pack_w<<<(8 * 16 * 32 + 255) / 256, 256, 0, stream>>>(W1, 256, 256, 256, 256, 16, wp + OFF1);
  pack_w<<<(8 * 16 * 32 + 255) / 256, 256, 0, stream>>>(W2, 256, 256, 256, 256, 16, wp + OFF2);
  pack_w<<<(8 * 16 * 32 + 255) / 256, 256, 0, stream>>>(W3, 256, 256, 256, 256, 16, wp + OFF3);
  pack_w<<<1, 256, 0, stream>>>(W4, 256, 256, 3, 3, 1, wp + OFF4);

  // 2304 blocks * 2 waves = 4608 waves -> exactly 4 tile-pairs per wave
  liif_fused<<<2304, 64, 0, stream>>>(x, coord, cell, lr,
                                      b0, b1, b2, b3, b4, wp, (float*)d_out);
}
